// GroupedQueryAttention_16088947491019
// MI455X (gfx1250) — matrix operand order repaired and checked by reading, not compiled, not run on hardware
//
#include <hip/hip_runtime.h>

// CDNA5 / gfx1250, wave32. Matrix math via V_WMMA_F32_16X16X32_F16.

typedef __attribute__((ext_vector_type(16))) _Float16 v16h;
typedef __attribute__((ext_vector_type(8)))  float    v8f;

union Frag { v16h v; uint4 q[2]; };

__device__ __forceinline__ v8f wmma_f16(const Frag& a, const Frag& b, v8f c) {
  // 8 args: (neg_a, A, neg_b, B, c_mod, C, reuse_a, reuse_b)
  return __builtin_amdgcn_wmma_f32_16x16x32_f16(false, a.v, false, b.v,
                                                (short)0, c, false, false);
}

// ---------------------------------------------------------------------------
// Elementwise helpers
// ---------------------------------------------------------------------------
__global__ void cvt_f16_kernel(const float* __restrict__ in,
                               _Float16* __restrict__ out, size_t n) {
  size_t i = (size_t)blockIdx.x * blockDim.x + threadIdx.x;
  if (i < n) out[i] = (_Float16)in[i];
}

// in: [R][C] f32 row-major  ->  out: [C][R] f16 row-major (out[o], o = c*R + r)
__global__ void transpose_cvt_kernel(const float* __restrict__ in,
                                     _Float16* __restrict__ out, int R, int C) {
  size_t o = (size_t)blockIdx.x * blockDim.x + threadIdx.x;
  if (o >= (size_t)R * C) return;
  int r = (int)(o % (size_t)R);
  int c = (int)(o / (size_t)R);
  out[o] = (_Float16)in[(size_t)r * C + c];
}

// vf: [B*T][G*256] f32  ->  vt: [B*G][256][T] f16  (K-contiguous columns for P@V)
__global__ void vtrans_kernel(const float* __restrict__ vf,
                              _Float16* __restrict__ vt, int T) {
  size_t o = (size_t)blockIdx.x * blockDim.x + threadIdx.x;
  size_t total = (size_t)8 * 256 * T;
  if (o >= total) return;
  int t  = (int)(o % (size_t)T);
  int d  = (int)((o / (size_t)T) % 256);
  int bg = (int)(o / ((size_t)T * 256));
  int b = bg >> 2, g = bg & 3;
  vt[o] = (_Float16)vf[(size_t)(b * T + t) * 1024 + g * 256 + d];
}

// Gemma RMSNorm over 256-wide head vectors; also rearranges to [B*numH][T][256].
__global__ void rmsnorm_kernel(const float* __restrict__ in,
                               const float* __restrict__ scale,
                               _Float16* __restrict__ out, int T, int numH) {
  const int row  = blockIdx.x;          // b*T + t
  const int h    = threadIdx.x >> 5;    // one wave per head
  const int lane = threadIdx.x & 31;
  const float* p = in + (size_t)row * numH * 256 + h * 256 + lane * 8;
  float v[8];
  float ss = 0.0f;
#pragma unroll
  for (int i = 0; i < 8; ++i) { v[i] = p[i]; ss += v[i] * v[i]; }
#pragma unroll
  for (int off = 1; off <= 16; off <<= 1) ss += __shfl_xor(ss, off, 32);
  const float rs = rsqrtf(ss * (1.0f / 256.0f) + 1e-6f);
  const int b = row / T, t = row % T;
  _Float16* op = out + ((size_t)(b * numH + h) * T + t) * 256 + lane * 8;
#pragma unroll
  for (int i = 0; i < 8; ++i)
    op[i] = (_Float16)(v[i] * rs * (1.0f + scale[lane * 8 + i]));
}

// ---------------------------------------------------------------------------
// WMMA GEMM:  C[M][N] f32 = A[M][K] f16 (row-major) @ BT[N][K] f16 (K-contig)
// Block = 256 threads = 8 waves; wave -> 16 rows, block -> 128x64 tile.
// ---------------------------------------------------------------------------
__global__ __launch_bounds__(256) void gemm_f16_nt(
    const _Float16* __restrict__ A, const _Float16* __restrict__ BT,
    float* __restrict__ C, int N, int K) {
  const int lane = threadIdx.x & 31;
  const int wave = threadIdx.x >> 5;
  const int m0   = blockIdx.y * 128 + wave * 16;
  const int n0   = blockIdx.x * 64;
  const int col  = lane & 15;
  const int half = lane >> 4;

  v8f acc[4];
#pragma unroll
  for (int j = 0; j < 4; ++j)
#pragma unroll
    for (int r = 0; r < 8; ++r) acc[j][r] = 0.0f;

  const _Float16* ap = A + (size_t)(m0 + col) * K + half * 8;
  for (int k0 = 0; k0 < K; k0 += 32) {
    Frag a;
    a.q[0] = *(const uint4*)(ap + k0);
    a.q[1] = *(const uint4*)(ap + k0 + 16);
    __builtin_prefetch((const void*)(ap + k0 + 64), 0, 3);  // global_prefetch_b8
#pragma unroll
    for (int j = 0; j < 4; ++j) {
      const _Float16* bp = BT + (size_t)(n0 + j * 16 + col) * K + k0 + half * 8;
      Frag b;
      b.q[0] = *(const uint4*)bp;
      b.q[1] = *(const uint4*)(bp + 16);
      acc[j] = wmma_f16(a, b, acc[j]);
    }
  }
#pragma unroll
  for (int j = 0; j < 4; ++j)
#pragma unroll
    for (int r = 0; r < 8; ++r)
      C[(size_t)(m0 + r + 8 * half) * N + n0 + j * 16 + col] = acc[j][r];
}

// ---------------------------------------------------------------------------
// Flash attention, sliding window 1024, one wave per 16-query tile.
// qn: [B*H][T][256], kn: [B*G][T][256], vt: [B*G][256][T], attn: [B*T][2048]
// ---------------------------------------------------------------------------
__global__ __launch_bounds__(32) void attn_kernel(
    const _Float16* __restrict__ qn, const _Float16* __restrict__ kn,
    const _Float16* __restrict__ vt, _Float16* __restrict__ attn) {
  const int T = 2048;
  const int lane = threadIdx.x & 31;
  const int col  = lane & 15;
  const int half = lane >> 4;
  const int t0 = blockIdx.x * 16;
  const int bh = blockIdx.y;                 // b*8 + h
  const int b = bh >> 3, h = bh & 7;
  const int bg = b * 4 + (h & 3);            // kv group = h % G (faithful mapping)

  const _Float16* qbase = qn + (size_t)bh * T * 256;
  const _Float16* kbase = kn + (size_t)bg * T * 256;
  const _Float16* vbase = vt + (size_t)bg * 256 * T;

  // Q tile as 8 A-fragments (Hd = 256 = 8 * 32)
  Frag qa[8];
#pragma unroll
  for (int c = 0; c < 8; ++c) {
    const _Float16* p = qbase + (size_t)(t0 + col) * 256 + c * 32 + half * 8;
    qa[c].q[0] = *(const uint4*)p;
    qa[c].q[1] = *(const uint4*)(p + 16);
  }

  v8f o[16];
  v8f m, l;
#pragma unroll
  for (int r = 0; r < 8; ++r) { m[r] = -1e30f; l[r] = 0.0f; }
#pragma unroll
  for (int j = 0; j < 16; ++j)
#pragma unroll
    for (int r = 0; r < 8; ++r) o[j][r] = 0.0f;

  __shared__ __align__(16) _Float16 Pl[16 * 32];  // C-layout -> A-layout bounce

  int kb_lo = t0 - 1023; if (kb_lo < 0) kb_lo = 0; kb_lo &= ~31;
  for (int kb = kb_lo; kb <= t0 + 15; kb += 32) {
    v8f s0, s1;
#pragma unroll
    for (int tsel = 0; tsel < 2; ++tsel) {
      v8f sc;
#pragma unroll
      for (int r = 0; r < 8; ++r) sc[r] = 0.0f;
      const int kcol = kb + tsel * 16 + col;   // absolute key for this lane's col
      const _Float16* kp = kbase + (size_t)kcol * 256 + half * 8;
#pragma unroll
      for (int c = 0; c < 8; ++c) {            // S += Q_chunk @ K_chunk^T
        Frag kf;
        kf.q[0] = *(const uint4*)(kp + c * 32);
        kf.q[1] = *(const uint4*)(kp + c * 32 + 16);
        sc = wmma_f16(qa[c], kf, sc);
      }
      v8f& s = tsel ? s1 : s0;
#pragma unroll
      for (int r = 0; r < 8; ++r) {            // scale + sliding causal mask
        int qabs = t0 + r + 8 * half;
        bool ok = (kcol <= qabs) && (kcol > qabs - 1024);
        s[r] = ok ? sc[r] * 0.0625f : -1e30f;
      }
    }
    // online softmax: row stats live in the same C-fragment layout
    v8f mnew, alpha;
#pragma unroll
    for (int r = 0; r < 8; ++r) {
      float rm = fmaxf(s0[r], s1[r]);
#pragma unroll
      for (int off = 1; off <= 8; off <<= 1)
        rm = fmaxf(rm, __shfl_xor(rm, off, 32));
      mnew[r]  = fmaxf(m[r], rm);
      alpha[r] = __expf(m[r] - mnew[r]);
      m[r]     = mnew[r];
    }
#pragma unroll
    for (int r = 0; r < 8; ++r) {
      s0[r] = __expf(s0[r] - mnew[r]);
      s1[r] = __expf(s1[r] - mnew[r]);
      float rs = s0[r] + s1[r];
#pragma unroll
      for (int off = 1; off <= 8; off <<= 1)
        rs += __shfl_xor(rs, off, 32);
      l[r] = l[r] * alpha[r] + rs;
    }
#pragma unroll
    for (int j = 0; j < 16; ++j)
#pragma unroll
      for (int r = 0; r < 8; ++r) o[j][r] *= alpha[r];

    // P (C-layout) -> LDS -> A-fragment
#pragma unroll
    for (int r = 0; r < 8; ++r) {
      Pl[(r + 8 * half) * 32 + col]      = (_Float16)s0[r];
      Pl[(r + 8 * half) * 32 + 16 + col] = (_Float16)s1[r];
    }
    __syncthreads();
    Frag pa;
    {
      const _Float16* pp = Pl + col * 32 + half * 8;
      pa.q[0] = *(const uint4*)pp;
      pa.q[1] = *(const uint4*)(pp + 16);
    }
    __syncthreads();
    // O += P @ V  (V pre-transposed: rows = head-dim, K-contiguous keys)
#pragma unroll
    for (int j = 0; j < 16; ++j) {
      const _Float16* vp = vbase + (size_t)(j * 16 + col) * T + kb + half * 8;
      Frag vfr;
      vfr.q[0] = *(const uint4*)vp;
      vfr.q[1] = *(const uint4*)(vp + 16);
      o[j] = wmma_f16(pa, vfr, o[j]);
    }
  }

  v8f rinv;
#pragma unroll
  for (int r = 0; r < 8; ++r) rinv[r] = __builtin_amdgcn_rcpf(l[r]);
#pragma unroll
  for (int j = 0; j < 16; ++j)
#pragma unroll
    for (int r = 0; r < 8; ++r) {
      size_t idx = (size_t)(b * T + t0 + r + 8 * half) * 2048
                 + h * 256 + j * 16 + col;
      attn[idx] = (_Float16)(o[j][r] * rinv[r]);
    }
}

// ---------------------------------------------------------------------------
extern "C" void kernel_launch(void* const* d_in, const int* in_sizes, int n_in,
                              void* d_out, int out_size, void* d_ws, size_t ws_size,
                              hipStream_t stream) {
  (void)in_sizes; (void)n_in; (void)out_size; (void)ws_size;
  const float* x  = (const float*)d_in[0];
  const float* Wq = (const float*)d_in[1];
  const float* Wk = (const float*)d_in[2];
  const float* Wv = (const float*)d_in[3];
  const float* Wo = (const float*)d_in[4];
  const float* qs = (const float*)d_in[5];
  const float* ks = (const float*)d_in[6];
  float* out = (float*)d_out;

  const int B = 2, T = 2048, D = 2560, H = 8, G = 4, Hd = 256;
  const int M  = B * T;        // 4096
  const int NQ = H * Hd;       // 2048
  const int NK = G * Hd;       // 1024

  char* ws = (char*)d_ws;
  size_t off = 0;
  auto alloc = [&](size_t bytes) -> void* {
    void* p = ws + off;
    off = (off + bytes + 255) & ~(size_t)255;
    return p;
  };
  _Float16* xh   = (_Float16*)alloc((size_t)M * D * 2);
  _Float16* WqT  = (_Float16*)alloc((size_t)NQ * D * 2);
  _Float16* WkT  = (_Float16*)alloc((size_t)NK * D * 2);
  _Float16* WvT  = (_Float16*)alloc((size_t)NK * D * 2);
  _Float16* WoT  = (_Float16*)alloc((size_t)D * NQ * 2);
  float*    qf   = (float*)alloc((size_t)M * NQ * 4);
  float*    kf   = (float*)alloc((size_t)M * NK * 4);
  float*    vf   = (float*)alloc((size_t)M * NK * 4);
  _Float16* qn   = (_Float16*)alloc((size_t)M * NQ * 2);
  _Float16* kn   = (_Float16*)alloc((size_t)M * NK * 2);
  _Float16* vt   = (_Float16*)alloc((size_t)M * NK * 2);
  _Float16* attn = (_Float16*)qf;   // qf is dead after rmsnorm; reuse for attn out

  // 1) x -> f16
  { size_t n = (size_t)M * D;
    cvt_f16_kernel<<<dim3((unsigned)((n + 255) / 256)), dim3(256), 0, stream>>>(x, xh, n); }
  // 2) weight transposes (f32 [K][N] -> f16 [N][K])
  { size_t n = (size_t)D * NQ;
    transpose_cvt_kernel<<<dim3((unsigned)((n + 255) / 256)), dim3(256), 0, stream>>>(Wq, WqT, D, NQ); }
  { size_t n = (size_t)D * NK;
    transpose_cvt_kernel<<<dim3((unsigned)((n + 255) / 256)), dim3(256), 0, stream>>>(Wk, WkT, D, NK); }
  { size_t n = (size_t)D * NK;
    transpose_cvt_kernel<<<dim3((unsigned)((n + 255) / 256)), dim3(256), 0, stream>>>(Wv, WvT, D, NK); }
  { size_t n = (size_t)NQ * D;
    transpose_cvt_kernel<<<dim3((unsigned)((n + 255) / 256)), dim3(256), 0, stream>>>(Wo, WoT, NQ, D); }
  // 3) projections (WMMA)
  gemm_f16_nt<<<dim3(NQ / 64, M / 128), dim3(256), 0, stream>>>(xh, WqT, qf, NQ, D);
  gemm_f16_nt<<<dim3(NK / 64, M / 128), dim3(256), 0, stream>>>(xh, WkT, kf, NK, D);
  gemm_f16_nt<<<dim3(NK / 64, M / 128), dim3(256), 0, stream>>>(xh, WvT, vf, NK, D);
  // 4) qk RMSNorm (+ rearrange to [B*heads][T][256] f16)
  rmsnorm_kernel<<<dim3(M), dim3(H * 32), 0, stream>>>(qf, qs, qn, T, H);
  rmsnorm_kernel<<<dim3(M), dim3(G * 32), 0, stream>>>(kf, ks, kn, T, G);
  // 5) V -> [B*G][256][T] f16
  { size_t n = (size_t)B * G * Hd * T;
    vtrans_kernel<<<dim3((unsigned)((n + 255) / 256)), dim3(256), 0, stream>>>(vf, vt, T); }
  // 6) sliding-window flash attention (WMMA)
  attn_kernel<<<dim3(T / 16, B * H), dim3(32), 0, stream>>>(qn, kn, vt, attn);
  // 7) output projection (WMMA) -> fp32 d_out
  gemm_f16_nt<<<dim3(D / 64, M / 128), dim3(256), 0, stream>>>(attn, WoT, out, D, NQ);
}
